// reactionMPNN_13228499272145
// MI455X (gfx1250) — matrix operand order repaired and checked by reading, not compiled, not run on hardware
//
#include <hip/hip_runtime.h>

// ---------------- problem constants (from reference) ----------------
#define N_NODES  65536      // B * NODES_PER_MOL
#define N_EDGES  131072
#define BATCH    2048
#define NODE_IN  64
#define EDGE_IN  16
#define EPAD     32         // edge feats padded to one k-step
#define H_DIM    300
#define HP       320        // H padded to 10 k-steps of 32 / 20 n-tiles of 16
#define DEPTH_L  3

#define LDSK     72         // padded LDS column stride (halves): kills bank conflicts
                            // (144B stride -> lane n hits bank (36n)%64, all distinct)

typedef __attribute__((ext_vector_type(16))) _Float16 v16h;
typedef __attribute__((ext_vector_type(8)))  float    v8f;
typedef __attribute__((ext_vector_type(4)))  float    v4f;

// -------------------------------------------------------------------
// Convert + transpose + pad a weight matrix W[K, 300] (f32, row stride 300)
// into Wt[HP cols][Kp] (f16, column-major, zero padded).
// -------------------------------------------------------------------
__global__ void prep_weight(const float* __restrict__ W,
                            _Float16* __restrict__ Wt, int K, int Kp) {
    int i = blockIdx.x * blockDim.x + threadIdx.x;
    int total = HP * Kp;
    if (i >= total) return;
    int c = i / Kp;
    int k = i - c * Kp;
    float v = (c < H_DIM && k < K) ? W[(size_t)k * H_DIM + c] : 0.0f;
    Wt[(size_t)c * Kp + k] = (_Float16)v;
}

// Convert f32 activations [rows, inC] -> f16 [rows, outC], zero padded.
__global__ void convert_a(const float* __restrict__ X,
                          _Float16* __restrict__ A,
                          int rows, int inC, int outC) {
    int i = blockIdx.x * blockDim.x + threadIdx.x;
    if (i >= rows * outC) return;
    int r = i / outC;
    int c = i - r * outC;
    A[i] = (c < inC) ? (_Float16)X[(size_t)r * inC + c] : (_Float16)0.0f;
}

// -------------------------------------------------------------------
// WMMA GEMM: out = act( A[rows,Ka](f16) x Wt[HP,Ka](f16,col-major) + bias )
//   - block: 256 threads = 8 waves arranged 4 (M) x 2 (N)
//   - each wave: 2 M-tiles (32 rows) x 10 N-tiles (160 cols)
//     -> B fragment reused across 2 WMMAs: 1 ds_load_b128 per WMMA
//   - acc: 2 x 10 x v8f = 160 VGPRs live across the K loop
//   - W staged through a 45KB padded LDS panel in K-chunks of 64
//   - emits f32 out (stride 300, cols<300) and/or f16 out (stride 320)
// -------------------------------------------------------------------
__global__ void __launch_bounds__(256, 1)
gemm_wmma(const _Float16* __restrict__ A,
          const _Float16* __restrict__ Wt,
          const float* __restrict__ bias,
          float* __restrict__ outF32,
          _Float16* __restrict__ outF16,
          int Ka, int reluFlag) {
    __shared__ _Float16 ldsW[HP * LDSK];   // 46080 B

    const int tid      = threadIdx.x;
    const int wave     = tid >> 5;
    const int lane     = tid & 31;
    const int laneHalf = lane & 15;
    const int laneHi   = lane >> 4;
    const int waveM    = wave & 3;             // 4 M strips of 32 rows
    const int waveN    = wave >> 2;            // 2 N halves of 160 cols
    const int rowBase  = blockIdx.x * 128 + waveM * 32;
    const int ntBase   = waveN * 10;

    v8f zero = {};
    v8f acc[2][10];
#pragma unroll
    for (int m = 0; m < 2; ++m)
#pragma unroll
        for (int t = 0; t < 10; ++t) acc[m][t] = zero;

    const _Float16* arow0 = A + (size_t)(rowBase + laneHalf) * Ka;
    const _Float16* arow1 = A + (size_t)(rowBase + 16 + laneHalf) * Ka;

    for (int kc = 0; kc < Ka; kc += 64) {
        int kk = Ka - kc;
        if (kk > 64) kk = 64;
        __syncthreads();
        // cooperative fill: contiguous 16B chunks (Wt already col-major)
        const int cpc  = kk >> 3;            // 16B chunks per column
        const int nvec = HP * cpc;
        for (int i = tid; i < nvec; i += 256) {
            int c = i / cpc;
            int q = i - c * cpc;
            *reinterpret_cast<uint4*>(&ldsW[c * LDSK + q * 8]) =
                *reinterpret_cast<const uint4*>(&Wt[(size_t)c * Ka + kc + q * 8]);
        }
        __syncthreads();

        for (int ks = 0; ks * 32 < kk; ++ks) {
            const int k0 = kc + ks * 32;
            // A fragments (16x32 f16): lane l<16 row l holds K {0..7,16..23},
            // lane l+16 holds K {8..15,24..31} -> two contiguous 16B loads each
            v16h afrag0, afrag1;
            {
                const _Float16* ap0 = arow0 + k0 + laneHi * 8;
                v4f* av = reinterpret_cast<v4f*>(&afrag0);
                av[0] = *reinterpret_cast<const v4f*>(ap0);
                av[1] = *reinterpret_cast<const v4f*>(ap0 + 16);
                const _Float16* ap1 = arow1 + k0 + laneHi * 8;
                v4f* av1 = reinterpret_cast<v4f*>(&afrag1);
                av1[0] = *reinterpret_cast<const v4f*>(ap1);
                av1[1] = *reinterpret_cast<const v4f*>(ap1 + 16);
            }
#pragma unroll
            for (int nt = 0; nt < 10; ++nt) {
                // B fragment (32x16): lanes 0-15 col n hold K {0..7,16..23},
                // lanes 16-31 hold K {8..15,24..31} -> two 16B LDS reads
                const int c = (ntBase + nt) * 16 + laneHalf;
                const _Float16* wp = &ldsW[c * LDSK + (k0 - kc) + laneHi * 8];
                v16h bfrag;
                v4f* bv = reinterpret_cast<v4f*>(&bfrag);
                bv[0] = *reinterpret_cast<const v4f*>(wp);
                bv[1] = *reinterpret_cast<const v4f*>(wp + 16);

                acc[0][nt] = __builtin_amdgcn_wmma_f32_16x16x32_f16(
                    false, afrag0, false, bfrag, (short)0, acc[0][nt],
                    false, false);
                acc[1][nt] = __builtin_amdgcn_wmma_f32_16x16x32_f16(
                    false, afrag1, false, bfrag, (short)0, acc[1][nt],
                    false, false);
            }
        }
    }

    // epilogue: VGPR j holds (lane<16 ? M=j : M=j+8), N = tile*16 + laneHalf
#pragma unroll
    for (int m = 0; m < 2; ++m) {
#pragma unroll
        for (int nt = 0; nt < 10; ++nt) {
            const int col = (ntBase + nt) * 16 + laneHalf;
            float bv = 0.0f;
            if (bias != nullptr && col < H_DIM) bv = bias[col];
#pragma unroll
            for (int j = 0; j < 8; ++j) {
                const int r = rowBase + m * 16 + laneHi * 8 + j;
                float v = acc[m][nt][j] + bv;
                if (reluFlag) v = fmaxf(v, 0.0f);
                if (outF32 != nullptr && col < H_DIM)
                    outF32[(size_t)r * H_DIM + col] = v;
                if (outF16 != nullptr)
                    outF16[(size_t)r * HP + col] = (_Float16)v;
            }
        }
    }
}

// m = relu(h[src] + eh); agg[dst] += m  (hardware f32 atomics into L2)
__global__ void edge_msg(const float* __restrict__ h,
                         const float* __restrict__ eh,
                         const int* __restrict__ src,
                         const int* __restrict__ dst,
                         float* __restrict__ agg) {
    const int e = blockIdx.x;
    const int c = threadIdx.x;
    if (c >= H_DIM) return;
    const int s = src[e];
    const int d = dst[e];
    float m = h[(size_t)s * H_DIM + c] + eh[(size_t)e * H_DIM + c];
    m = fmaxf(m, 0.0f);
    __hip_atomic_fetch_add(&agg[(size_t)d * H_DIM + c], m,
                           __ATOMIC_RELAXED, __HIP_MEMORY_SCOPE_AGENT);
}

// z(f16, [N,HP]) = h + agg for cols < 300, else 0
__global__ void combine_z(const float* __restrict__ h,
                          const float* __restrict__ agg,
                          _Float16* __restrict__ z) {
    int i = blockIdx.x * blockDim.x + threadIdx.x;
    if (i >= N_NODES * HP) return;
    int r = i / HP;
    int c = i - r * HP;
    _Float16 v = (_Float16)0.0f;
    if (c < H_DIM)
        v = (_Float16)(h[(size_t)r * H_DIM + c] + agg[(size_t)r * H_DIM + c]);
    z[i] = v;
}

// seg = repeat(arange(B), 32): pooling = sum of 32 consecutive rows
__global__ void pool_sum(const float* __restrict__ h,
                         float* __restrict__ pooled) {
    int i = blockIdx.x * blockDim.x + threadIdx.x;
    if (i >= BATCH * H_DIM) return;
    int b = i / H_DIM;
    int c = i - b * H_DIM;
    const float* p = h + (size_t)b * 32 * H_DIM + c;
    float s = 0.0f;
#pragma unroll
    for (int n = 0; n < 32; ++n) s += p[(size_t)n * H_DIM];
    pooled[i] = s;
}

// out = [reactants - products ; reactants ; products]
__global__ void final_combine(const float* __restrict__ pr1,
                              const float* __restrict__ pr2,
                              const float* __restrict__ pp1,
                              float* __restrict__ out) {
    const int PB = BATCH * H_DIM;
    int i = blockIdx.x * blockDim.x + threadIdx.x;
    if (i >= PB) return;
    float rs = pr1[i] + pr2[i];
    float p  = pp1[i];
    out[i]          = rs - p;
    out[PB + i]     = rs;
    out[2 * PB + i] = p;
}

// -------------------------------------------------------------------
extern "C" void kernel_launch(void* const* d_in, const int* in_sizes, int n_in,
                              void* d_out, int out_size, void* d_ws, size_t ws_size,
                              hipStream_t stream) {
    const float* Wn = (const float*)d_in[15];
    const float* bn = (const float*)d_in[16];
    const float* We = (const float*)d_in[17];
    const float* be = (const float*)d_in[18];
    const float* W1 = (const float*)d_in[19];
    const float* b1 = (const float*)d_in[20];
    const float* W2 = (const float*)d_in[21];
    const float* b2 = (const float*)d_in[22];

    char* ws = (char*)d_ws;
    size_t off = 0;
    auto take = [&](size_t bytes) -> void* {
        void* p = ws + off;
        off = (off + bytes + 255) & ~(size_t)255;
        return p;
    };

    _Float16* wnT = (_Float16*)take((size_t)HP * NODE_IN * 2);
    _Float16* weT = (_Float16*)take((size_t)HP * EPAD * 2);
    _Float16* w1T[DEPTH_L];
    _Float16* w2T[DEPTH_L];
    for (int l = 0; l < DEPTH_L; ++l) w1T[l] = (_Float16*)take((size_t)HP * HP * 2);
    for (int l = 0; l < DEPTH_L; ++l) w2T[l] = (_Float16*)take((size_t)HP * HP * 2);
    _Float16* bufA = (_Float16*)take((size_t)N_NODES * HP * 2); // x16 / e16 / z16
    _Float16* bufT = (_Float16*)take((size_t)N_NODES * HP * 2); // MLP hidden
    float* hbuf    = (float*)take((size_t)N_NODES * H_DIM * 4);
    float* ehbuf   = (float*)take((size_t)N_EDGES * H_DIM * 4);
    float* agg     = (float*)take((size_t)N_NODES * H_DIM * 4);
    float* pooled[3];
    for (int g = 0; g < 3; ++g) pooled[g] = (float*)take((size_t)BATCH * H_DIM * 4);

    // ---- one-time weight convert/transpose/pad (tiny) ----
    {
        int tot = HP * NODE_IN;
        prep_weight<<<(tot + 255) / 256, 256, 0, stream>>>(Wn, wnT, NODE_IN, NODE_IN);
        tot = HP * EPAD;
        prep_weight<<<(tot + 255) / 256, 256, 0, stream>>>(We, weT, EDGE_IN, EPAD);
        tot = HP * HP;
        for (int l = 0; l < DEPTH_L; ++l) {
            prep_weight<<<(tot + 255) / 256, 256, 0, stream>>>(
                W1 + (size_t)l * H_DIM * H_DIM, w1T[l], H_DIM, HP);
            prep_weight<<<(tot + 255) / 256, 256, 0, stream>>>(
                W2 + (size_t)l * H_DIM * H_DIM, w2T[l], H_DIM, HP);
        }
    }

    for (int g = 0; g < 3; ++g) {
        const float* x  = (const float*)d_in[5 * g + 0];
        const float* e  = (const float*)d_in[5 * g + 1];
        const int* src  = (const int*)d_in[5 * g + 2];
        const int* dst  = (const int*)d_in[5 * g + 3];

        // h0 = relu(x @ Wn + bn)
        int tot = N_NODES * NODE_IN;
        convert_a<<<(tot + 255) / 256, 256, 0, stream>>>(x, bufA, N_NODES, NODE_IN, NODE_IN);
        gemm_wmma<<<N_NODES / 128, 256, 0, stream>>>(
            bufA, wnT, bn, hbuf, nullptr, NODE_IN, 1);

        // eh = e @ We + be
        tot = N_EDGES * EPAD;
        convert_a<<<(tot + 255) / 256, 256, 0, stream>>>(e, bufA, N_EDGES, EDGE_IN, EPAD);
        gemm_wmma<<<N_EDGES / 128, 256, 0, stream>>>(
            bufA, weT, be, ehbuf, nullptr, EPAD, 0);

        for (int l = 0; l < DEPTH_L; ++l) {
            hipMemsetAsync(agg, 0, (size_t)N_NODES * H_DIM * 4, stream);
            edge_msg<<<N_EDGES, 320, 0, stream>>>(hbuf, ehbuf, src, dst, agg);

            tot = N_NODES * HP;
            combine_z<<<(tot + 255) / 256, 256, 0, stream>>>(hbuf, agg, bufA);

            // t = relu(z @ W1 + b1)   (f16 out, stride HP)
            gemm_wmma<<<N_NODES / 128, 256, 0, stream>>>(
                bufA, w1T[l], b1 + (size_t)l * H_DIM, nullptr, bufT, HP, 1);
            // h = t @ W2 + b2 (+relu between layers)   (f32 out, stride 300)
            gemm_wmma<<<N_NODES / 128, 256, 0, stream>>>(
                bufT, w2T[l], b2 + (size_t)l * H_DIM, hbuf, nullptr, HP,
                (l < DEPTH_L - 1) ? 1 : 0);
        }

        tot = BATCH * H_DIM;
        pool_sum<<<(tot + 255) / 256, 256, 0, stream>>>(hbuf, pooled[g]);
    }

    final_combine<<<(BATCH * H_DIM + 255) / 256, 256, 0, stream>>>(
        pooled[0], pooled[1], pooled[2], (float*)d_out);
}
